// GAT0tmasterb_33200097198300
// MI455X (gfx1250) — matrix operand order repaired and checked by reading, not compiled, not run on hardware
//
#include <hip/hip_runtime.h>
#include <hip/hip_bf16.h>
#include <stdint.h>

#define NEG_SLOPE 0.2f
#define N_NODES 20000
#define N_EDGES 160000
#define N_GRAPHS 100

typedef __attribute__((ext_vector_type(16))) _Float16 v16h;
typedef __attribute__((ext_vector_type(8)))  float    v8f;

// ---------------------------------------------------------------------------
// CAS-based float atomic max (logits can be negative; int-trick unsafe).
// ---------------------------------------------------------------------------
__device__ inline void atomicMaxF(float* addr, float val) {
  unsigned int* ua = (unsigned int*)addr;
  unsigned int old = __hip_atomic_load(ua, __ATOMIC_RELAXED, __HIP_MEMORY_SCOPE_AGENT);
  while (true) {
    float f = __uint_as_float(old);
    if (f >= val) break;
    unsigned int assumed = old;
    old = atomicCAS(ua, assumed, __float_as_uint(val));
    if (old == assumed) break;
  }
}

// Pack two contiguous 8-float runs (k = 8*half+[0..7], k = 16+8*half+[0..7])
// into the CDNA5 16-bit A fragment order with optional input ReLU.
template<bool RELU>
__device__ inline v16h pack_frag(float4 c0, float4 c1, float4 c2, float4 c3) {
  float t[16] = {c0.x, c0.y, c0.z, c0.w, c1.x, c1.y, c1.z, c1.w,
                 c2.x, c2.y, c2.z, c2.w, c3.x, c3.y, c3.z, c3.w};
  v16h r;
#pragma unroll
  for (int i = 0; i < 16; ++i) {
    float v = t[i];
    if (RELU) v = v > 0.f ? v : 0.f;
    r[i] = (_Float16)v;
  }
  return r;
}

// ---------------------------------------------------------------------------
// WMMA GEMM: C[M,N] = (relu?)A[M,K] @ B[K,N]; f32 in/out, f16 matrix core.
// 2x2 micro-tile per wave (32m x 32n), block = 4 waves -> 32m x 128n.
// N,K are compile-time: all strides fold to immediates (no 64-bit mul chains,
// no spills). Requires M%32==0, N%128==0, K%32==0 (true for all layers here).
// ---------------------------------------------------------------------------
template<bool RELU_IN, int N, int K>
__global__ __launch_bounds__(128, 1)
void gemm_wmma_f16(const float* __restrict__ A,
                   const float* __restrict__ B,
                   float* __restrict__ C, int M) {
  const int wave = threadIdx.x >> 5;
  const int lane = threadIdx.x & 31;
  const int half = lane >> 4;
  const int l15  = lane & 15;
  const int m0   = blockIdx.x * 32;
  const int n0   = (blockIdx.y * 4 + wave) * 32;

  v8f acc00 = {}, acc01 = {}, acc10 = {}, acc11 = {};

  // Per-lane base pointers; all loop offsets become constants.
  const float4* pa0 = (const float4*)(A + (m0 + l15) * K + (half << 3));
  const float4* pa1 = (const float4*)(A + (m0 + 16 + l15) * K + (half << 3));
  const float*  pb  = B + (half << 3) * N + n0 + l15;

  for (int kb = 0; kb < K; kb += 32) {
    const int q = kb >> 2;  // float4 index for this k-block
    // ---- A fragments: two contiguous 8-float runs per lane -> 4x b128 each
    v16h a0 = pack_frag<RELU_IN>(pa0[q], pa0[q + 1], pa0[q + 4], pa0[q + 5]);
    v16h a1 = pack_frag<RELU_IN>(pa1[q], pa1[q + 1], pa1[q + 4], pa1[q + 5]);

    // ---- B fragments: element i -> k = kb + 8*half + i (+8 for i>=8), lane-coalesced cols
    v16h b0, b1;
#pragma unroll
    for (int i = 0; i < 16; ++i) {
      const float* p = pb + (kb + i + ((i >> 3) << 3)) * N;
      b0[i] = (_Float16)p[0];
      b1[i] = (_Float16)p[16];
    }

    acc00 = __builtin_amdgcn_wmma_f32_16x16x32_f16(false, a0, false, b0,
                                                   (short)0, acc00, false, false);
    acc01 = __builtin_amdgcn_wmma_f32_16x16x32_f16(false, a0, false, b1,
                                                   (short)0, acc01, false, false);
    acc10 = __builtin_amdgcn_wmma_f32_16x16x32_f16(false, a1, false, b0,
                                                   (short)0, acc10, false, false);
    acc11 = __builtin_amdgcn_wmma_f32_16x16x32_f16(false, a1, false, b1,
                                                   (short)0, acc11, false, false);
  }

  // D: VGPR r holds row 8*half + r of its 16x16 tile, col = lane&15
  float* c0 = C + (m0 + (half << 3)) * N + n0 + l15;
#pragma unroll
  for (int r = 0; r < 8; ++r) {
    c0[r * N]             = acc00[r];
    c0[r * N + 16]        = acc01[r];
    c0[(16 + r) * N]      = acc10[r];
    c0[(16 + r) * N + 16] = acc11[r];
  }
}

// ---------------------------------------------------------------------------
// Fills
// ---------------------------------------------------------------------------
__global__ void fill_f32(float* __restrict__ p, float v, int n) {
  int i = blockIdx.x * blockDim.x + threadIdx.x;
  if (i < n) p[i] = v;
}
__global__ void init_bias(float* __restrict__ out, const float* __restrict__ b,
                          int N, int F) {
  int i = blockIdx.x * blockDim.x + threadIdx.x;
  if (i < N * F) out[i] = b[i % F];
}

// ---------------------------------------------------------------------------
// Edge pass 1: per-edge attention logits + segment max over dst.
// Wave per edge, lanes stride channels. We (8xF) + att (F) staged in LDS.
// logit_eh = att_h . leaky_relu(xl[src] + xr[dst] + edge_attr@We)
// ---------------------------------------------------------------------------
template<bool HAS_E>
__global__ void edge_logit_max(const float* __restrict__ xl,
                               const float* __restrict__ xr,
                               const float* __restrict__ edge_attr,
                               const float* __restrict__ We,
                               const float* __restrict__ att,
                               const int* __restrict__ ei,
                               float* __restrict__ logit,
                               float* __restrict__ mx,
                               int E, int F, int H, int cshift) {
  extern __shared__ float lds[];
  float* lWe  = lds;
  float* lAtt = HAS_E ? (lds + 8 * F) : lds;
  const int ldsTot = HAS_E ? 9 * F : F;
  for (int i = threadIdx.x; i < ldsTot; i += blockDim.x) {
    if (HAS_E && i < 8 * F) lWe[i] = We[i];
    else                    lAtt[HAS_E ? (i - 8 * F) : i] = att[HAS_E ? (i - 8 * F) : i];
  }
  __syncthreads();

  const int lane = threadIdx.x & 31;
  const int wave = threadIdx.x >> 5;
  const int wpb  = blockDim.x >> 5;
  int e = blockIdx.x * wpb + wave;
  if (e >= E) return;

  const int src = ei[e];
  const int dst = ei[E + e];
  float ea[8];
  if (HAS_E) {
#pragma unroll
    for (int j = 0; j < 8; ++j) ea[j] = edge_attr[(size_t)e * 8 + j];
  }
  const float* pl = xl + (size_t)src * F;
  const float* pr = xr + (size_t)dst * F;
  float acc[6] = {0.f, 0.f, 0.f, 0.f, 0.f, 0.f};
  for (int c = lane; c < F; c += 32) {
    float mv = pl[c] + pr[c];
    if (HAS_E) {
      float s = 0.f;
#pragma unroll
      for (int j = 0; j < 8; ++j) s += ea[j] * lWe[j * F + c];
      mv += s;
    }
    mv = mv > 0.f ? mv : NEG_SLOPE * mv;
    acc[c >> cshift] += lAtt[c] * mv;
  }
  for (int h = 0; h < H; ++h) {
    float v = acc[h];
#pragma unroll
    for (int o = 16; o > 0; o >>= 1) v += __shfl_xor(v, o, 32);
    if (lane == 0) {
      logit[(size_t)e * H + h] = v;
      atomicMaxF(mx + (size_t)dst * H + h, v);
    }
  }
}

// ---------------------------------------------------------------------------
// Edge pass 2: ex = exp(logit - mx[dst]); den[dst] += ex (in-place overwrite)
// ---------------------------------------------------------------------------
__global__ void edge_exp_den(const int* __restrict__ ei,
                             float* __restrict__ lg,        // logit in, ex out
                             const float* __restrict__ mx,
                             float* __restrict__ den, int E, int H) {
  int t = blockIdx.x * blockDim.x + threadIdx.x;
  if (t >= E * H) return;
  int e = t / H, h = t - e * H;
  int dst = ei[E + e];
  float ex = __expf(lg[t] - mx[(size_t)dst * H + h]);
  lg[t] = ex;
  atomicAdd(den + (size_t)dst * H + h, ex);
}

// ---------------------------------------------------------------------------
// Edge pass 3: out[dst] += alpha_h * xl[src] (wave/edge, coalesced atomics)
// ---------------------------------------------------------------------------
__global__ void edge_scatter(const float* __restrict__ xl,
                             const float* __restrict__ ex,
                             const float* __restrict__ den,
                             const int* __restrict__ ei,
                             float* __restrict__ out,
                             int E, int F, int H, int cshift) {
  const int lane = threadIdx.x & 31;
  const int wave = threadIdx.x >> 5;
  const int wpb  = blockDim.x >> 5;
  int e = blockIdx.x * wpb + wave;
  if (e >= E) return;
  const int src = ei[e];
  const int dst = ei[E + e];
  float alpha[6];
  for (int h = 0; h < H; ++h)
    alpha[h] = ex[(size_t)e * H + h] / (den[(size_t)dst * H + h] + 1e-16f);
  const float* pl = xl + (size_t)src * F;
  float* po = out + (size_t)dst * F;
  for (int c = lane; c < F; c += 32)
    atomicAdd(po + c, alpha[c >> cshift] * pl[c]);
}

// ---------------------------------------------------------------------------
// Head: master = h3[cumsum(n_nodes)-1]; z = relu(master@Wfc1+b); out = z@Wfc2+b
// One block (64 threads) per graph.
// ---------------------------------------------------------------------------
__global__ void master_head(const float* __restrict__ h3,
                            const int* __restrict__ n_nodes,
                            const float* __restrict__ Wfc1,
                            const float* __restrict__ bfc1,
                            const float* __restrict__ Wfc2,
                            const float* __restrict__ bfc2,
                            float* __restrict__ out) {
  __shared__ float z[64];
  __shared__ int sidx;
  int g = blockIdx.x;
  if (threadIdx.x == 0) {
    int s = 0;
    for (int i = 0; i <= g; ++i) s += n_nodes[i];
    sidx = s - 1;
  }
  __syncthreads();
  const float* row = h3 + (size_t)sidx * 384;
  int o = threadIdx.x;  // 64 threads
  float s = bfc1[o];
  for (int k = 0; k < 384; ++k) s += row[k] * Wfc1[k * 64 + o];
  z[o] = s > 0.f ? s : 0.f;
  __syncthreads();
  if (threadIdx.x == 0) {
    float r = bfc2[0];
    for (int k = 0; k < 64; ++k) r += z[k] * Wfc2[k];
    out[g] = r;
  }
}

// ---------------------------------------------------------------------------
extern "C" void kernel_launch(void* const* d_in, const int* in_sizes, int n_in,
                              void* d_out, int out_size, void* d_ws, size_t ws_size,
                              hipStream_t stream) {
  const float* x     = (const float*)d_in[0];
  const float* eattr = (const float*)d_in[1];
  const float* Wl1   = (const float*)d_in[2];
  const float* Wr1   = (const float*)d_in[3];
  const float* We1   = (const float*)d_in[4];
  const float* att1  = (const float*)d_in[5];
  const float* b1    = (const float*)d_in[6];
  const float* Wl2   = (const float*)d_in[7];
  const float* Wr2   = (const float*)d_in[8];
  const float* We2   = (const float*)d_in[9];
  const float* att2  = (const float*)d_in[10];
  const float* b2    = (const float*)d_in[11];
  const float* Wlp   = (const float*)d_in[12];
  const float* Wrp   = (const float*)d_in[13];
  const float* attp  = (const float*)d_in[14];
  const float* bp    = (const float*)d_in[15];
  const float* Wfc1  = (const float*)d_in[16];
  const float* bfc1  = (const float*)d_in[17];
  const float* Wfc2  = (const float*)d_in[18];
  const float* bfc2  = (const float*)d_in[19];
  const int*   ei    = (const int*)d_in[20];
  const int*   eim   = (const int*)d_in[21];
  const int*   nn    = (const int*)d_in[22];
  float* out = (float*)d_out;

  // Workspace layout (floats). Peak ~312 MB.
  float* ws   = (float*)d_ws;
  float* xl1  = ws;                       // 20000*1024  (reused as xlp)
  float* xr1  = ws + 20480000;            // 20000*1024  (reused as xrp)
  float* h1   = ws + 40960000;            // 20000*1024  (reused as h3: 20000*384)
  float* xl2  = ws + 61440000;            // 20000*256
  float* xr2  = ws + 66560000;            // 20000*256
  float* h2   = ws + 71680000;            // 20000*256
  float* lg   = ws + 76800000;            // 160000*6 (logit -> ex)
  float* mx   = ws + 77760000;            // 20000*6
  float* den  = ws + 77880000;            // 20000*6

  const int E = N_EDGES, Nn = N_NODES;
  const dim3 blk128(128), blk256(256), blk64(64);
  const int edgeBlocks = (E + 7) / 8;     // 8 waves/block, wave per edge

  // ===== Layer 1: F=1024, H=4, Cph=256 (shift 8) =====
  gemm_wmma_f16<false, 1024, 64><<<dim3(Nn/32, 1024/128), blk128, 0, stream>>>(x, Wl1, xl1, Nn);
  gemm_wmma_f16<false, 1024, 64><<<dim3(Nn/32, 1024/128), blk128, 0, stream>>>(x, Wr1, xr1, Nn);
  fill_f32<<<(Nn*4 + 255)/256, blk256, 0, stream>>>(mx, -1e30f, Nn*4);
  fill_f32<<<(Nn*4 + 255)/256, blk256, 0, stream>>>(den, 0.f, Nn*4);
  init_bias<<<((size_t)Nn*1024 + 255)/256, blk256, 0, stream>>>(h1, b1, Nn, 1024);
  edge_logit_max<true><<<edgeBlocks, blk256, 9*1024*sizeof(float), stream>>>(
      xl1, xr1, eattr, We1, att1, ei, lg, mx, E, 1024, 4, 8);
  edge_exp_den<<<(E*4 + 255)/256, blk256, 0, stream>>>(ei, lg, mx, den, E, 4);
  edge_scatter<<<edgeBlocks, blk256, 0, stream>>>(xl1, lg, den, ei, h1, E, 1024, 4, 8);

  // ===== Layer 2: F=256, H=4, Cph=64 (shift 6); relu(h1) fused into GEMM =====
  gemm_wmma_f16<true, 256, 1024><<<dim3(Nn/32, 256/128), blk128, 0, stream>>>(h1, Wl2, xl2, Nn);
  gemm_wmma_f16<true, 256, 1024><<<dim3(Nn/32, 256/128), blk128, 0, stream>>>(h1, Wr2, xr2, Nn);
  fill_f32<<<(Nn*4 + 255)/256, blk256, 0, stream>>>(mx, -1e30f, Nn*4);
  fill_f32<<<(Nn*4 + 255)/256, blk256, 0, stream>>>(den, 0.f, Nn*4);
  init_bias<<<((size_t)Nn*256 + 255)/256, blk256, 0, stream>>>(h2, b2, Nn, 256);
  edge_logit_max<true><<<edgeBlocks, blk256, 9*256*sizeof(float), stream>>>(
      xl2, xr2, eattr, We2, att2, ei, lg, mx, E, 256, 4, 6);
  edge_exp_den<<<(E*4 + 255)/256, blk256, 0, stream>>>(ei, lg, mx, den, E, 4);
  edge_scatter<<<edgeBlocks, blk256, 0, stream>>>(xl2, lg, den, ei, h2, E, 256, 4, 6);

  // ===== Layer p: F=384, H=6, Cph=64 (shift 6); master edge index, no eattr ==
  float* xlp = xl1; float* xrp = xr1; float* h3 = h1;   // buffer reuse
  gemm_wmma_f16<true, 384, 256><<<dim3(Nn/32, 384/128), blk128, 0, stream>>>(h2, Wlp, xlp, Nn);
  gemm_wmma_f16<true, 384, 256><<<dim3(Nn/32, 384/128), blk128, 0, stream>>>(h2, Wrp, xrp, Nn);
  fill_f32<<<(Nn*6 + 255)/256, blk256, 0, stream>>>(mx, -1e30f, Nn*6);
  fill_f32<<<(Nn*6 + 255)/256, blk256, 0, stream>>>(den, 0.f, Nn*6);
  init_bias<<<((size_t)Nn*384 + 255)/256, blk256, 0, stream>>>(h3, bp, Nn, 384);
  edge_logit_max<false><<<edgeBlocks, blk256, 384*sizeof(float), stream>>>(
      xlp, xrp, nullptr, nullptr, attp, eim, lg, mx, E, 384, 6, 6);
  edge_exp_den<<<(E*6 + 255)/256, blk256, 0, stream>>>(eim, lg, mx, den, E, 6);
  edge_scatter<<<edgeBlocks, blk256, 0, stream>>>(xlp, lg, den, eim, h3, E, 384, 6, 6);

  // ===== Head =====
  master_head<<<N_GRAPHS, blk64, 0, stream>>>(h3, nn, Wfc1, bfc1, Wfc2, bfc2, out);
}
